// RNN_20564303413717
// MI455X (gfx1250) — matrix operand order repaired and checked by reading, not compiled, not run on hardware
//
#include <hip/hip_runtime.h>

// ---------------- problem constants ----------------
constexpr int SEQ    = 2048;
constexpr int BATCH  = 128;
constexpr int VOCAB  = 1024;
constexpr int HIDDEN = 256;

// ---------------- WMMA types ----------------
typedef __attribute__((ext_vector_type(16))) __bf16          v16bf;
typedef __attribute__((ext_vector_type(8)))  float           v8f;
typedef __attribute__((ext_vector_type(8)))  unsigned short  us8;
typedef __attribute__((ext_vector_type(4)))  float           f4;

union Frag {
    v16bf          bf;
    us8            h[2];
    unsigned short u[16];
};

// f32 -> bf16 (native hardware conversion, RNE)
__device__ __forceinline__ unsigned short f2bf(float f) {
    __bf16 b = (__bf16)f;
    return __builtin_bit_cast(unsigned short, b);
}

__device__ __forceinline__ float fast_tanh(float x) {
#if __has_builtin(__builtin_amdgcn_tanhf)
    return __builtin_amdgcn_tanhf(x);
#else
    return tanhf(x);
#endif
}

// ============================================================
// Phase 0: convert weight matrices f32 -> bf16 (once, tiny)
// ============================================================
__global__ void cvt_weights_kernel(const float* __restrict__ Wih,
                                   const float* __restrict__ Whh,
                                   const float* __restrict__ Wfc,
                                   unsigned short* __restrict__ oih,
                                   unsigned short* __restrict__ ohh,
                                   unsigned short* __restrict__ ofc) {
    const int N_IH = HIDDEN * VOCAB;
    const int N_HH = HIDDEN * HIDDEN;
    const int N_FC = VOCAB * HIDDEN;
    int i = blockIdx.x * blockDim.x + threadIdx.x;
    if (i < N_IH)                      oih[i] = f2bf(Wih[i]);
    else if (i < N_IH + N_HH)          ohh[i - N_IH] = f2bf(Whh[i - N_IH]);
    else if (i < N_IH + N_HH + N_FC)   ofc[i - N_IH - N_HH] = f2bf(Wfc[i - N_IH - N_HH]);
}

// ============================================================
// Phase 1: xi[m,n] = sum_k x[m,k] * W_ih[n,k] + (b_ih+b_hh)[n]
//   M = SEQ*BATCH, K = VOCAB, N = HIDDEN.  One 16x16 tile per wave.
// ============================================================
__global__ void xi_gemm_kernel(const float* __restrict__ x,
                               const unsigned short* __restrict__ Wih,
                               const float* __restrict__ b_ih,
                               const float* __restrict__ b_hh,
                               float* __restrict__ xi) {
    const int lane = threadIdx.x & 31;
    const int wave = threadIdx.x >> 5;          // 0..15 -> N tile
    const int m0   = blockIdx.x * 16;
    const int n0   = wave * 16;
    const int ah   = (lane >> 4) << 3;          // A half: +0 / +8
    const int bh   = (lane >> 4) << 4;          // B half: +0 / +16
    const int col  = lane & 15;

    const float*          arow = x   + (m0 + col) * VOCAB;
    const unsigned short* brow = Wih + (n0 + col) * VOCAB;

    v8f acc = {};
#pragma unroll 4
    for (int kk = 0; kk < VOCAB; kk += 32) {
        // A fragment: load f32, native-convert to bf16 in-register
        Frag fa;
        f4 a0 = *(const f4*)(arow + kk + ah);
        f4 a1 = *(const f4*)(arow + kk + ah + 4);
        f4 a2 = *(const f4*)(arow + kk + ah + 16);
        f4 a3 = *(const f4*)(arow + kk + ah + 20);
#pragma unroll
        for (int i = 0; i < 4; ++i) {
            fa.bf[i]      = (__bf16)a0[i];
            fa.bf[4 + i]  = (__bf16)a1[i];
            fa.bf[8 + i]  = (__bf16)a2[i];
            fa.bf[12 + i] = (__bf16)a3[i];
        }
        // B fragment: bf16 from W_ih row, k = ah+0..7 and ah+16..23 (same order as A)
        Frag fb;
        const us8* qb = (const us8*)(brow + kk + ah);
        fb.h[0] = qb[0];
        fb.h[1] = qb[2];
        acc = __builtin_amdgcn_wmma_f32_16x16x32_bf16(false, fa.bf, false, fb.bf,
                                                      (short)0, acc, false, false);
    }

    const float bias = b_ih[n0 + col] + b_hh[n0 + col];
#pragma unroll
    for (int r = 0; r < 8; ++r)
        xi[(m0 + r + ah) * HIDDEN + n0 + col] = acc[r] + bias;
}

// ============================================================
// Phase 2: sequential recurrence, batch-partitioned.
//   8 blocks x 16 waves; block owns 16 batch rows, wave owns 16 hidden cols.
//   h state in double-buffered LDS, one barrier per timestep.
//   W_hh fragments held in registers across all 2048 steps.
// ============================================================
__global__ void rnn_steps_kernel(const float* __restrict__ xi,
                                 const unsigned short* __restrict__ Whh,
                                 const float* __restrict__ h0,
                                 unsigned short* __restrict__ hs) {
    constexpr int LDH = HIDDEN + 16;            // 272 shorts = 544 B rows (16B aligned)
    __shared__ unsigned short hbuf[2][16][LDH];

    const int lane = threadIdx.x & 31;
    const int wave = threadIdx.x >> 5;          // 0..15
    const int b0   = blockIdx.x * 16;
    const int n0   = wave * 16;
    const int ah   = (lane >> 4) << 3;
    const int bh   = (lane >> 4) << 4;
    const int col  = lane & 15;

    // init h_{-1} = h0 (f32 -> bf16) into buffer 0
    for (int i = threadIdx.x; i < 16 * HIDDEN; i += blockDim.x) {
        int r = i >> 8, c = i & (HIDDEN - 1);
        hbuf[0][r][c] = f2bf(h0[(b0 + r) * HIDDEN + c]);
    }
    __syncthreads();

    // Hoist all 8 W_hh B-fragments into registers (loop-invariant, 64 VGPRs)
    const unsigned short* brow = Whh + (n0 + col) * HIDDEN;
    v16bf fb[8];
#pragma unroll
    for (int k8 = 0; k8 < 8; ++k8) {
        Frag t;
        const us8* qb = (const us8*)(brow + k8 * 32 + ah);
        t.h[0] = qb[0];
        t.h[1] = qb[2];
        fb[k8] = t.bf;
    }

    // cooperative hs-streaming coordinates: one 16B chunk per thread per step
    const int srow = threadIdx.x >> 5;          // 0..15
    const int scol = (threadIdx.x & 31) * 8;    // 0..248

    for (int s = 0; s < SEQ; ++s) {
        const int cur  = s & 1, nxt = cur ^ 1;
        const int base = (s * BATCH + b0) * HIDDEN;

        // prefetch next step's xi tile into cache while this step computes
        if (s + 1 < SEQ)
            __builtin_prefetch(&xi[base + BATCH * HIDDEN + ah * HIDDEN + n0 + col], 0, 0);

        // C = xi tile (f32)
        v8f acc;
#pragma unroll
        for (int r = 0; r < 8; ++r)
            acc[r] = xi[base + (r + ah) * HIDDEN + n0 + col];

        // acc += h_{s-1} · W_hh^T  (8 accumulating WMMAs, K = 256)
#pragma unroll
        for (int k8 = 0; k8 < 8; ++k8) {
            Frag fa;
            const unsigned short* ap = &hbuf[cur][col][k8 * 32 + ah];
            fa.h[0] = *(const us8*)ap;
            fa.h[1] = *(const us8*)(ap + 16);
            acc = __builtin_amdgcn_wmma_f32_16x16x32_bf16(false, fa.bf, false, fb[k8],
                                                          (short)0, acc, false, false);
        }

        // tanh, publish new h to LDS ping-pong buffer
#pragma unroll
        for (int r = 0; r < 8; ++r)
            hbuf[nxt][r + ah][n0 + col] = f2bf(fast_tanh(acc[r]));
        __syncthreads();

        // coalesced stream of the fresh h tile to hs (one b128 store per thread);
        // safe with a single barrier: step s+2's writes to this buffer are ordered
        // after these reads by the barrier at the end of step s+1.
        *(us8*)&hs[base + srow * HIDDEN + scol] = *(const us8*)&hbuf[nxt][srow][scol];
    }
}

// ============================================================
// Phase 3: out[m,n] = sum_k hs[m,k] * W_fc[n,k] + b_fc[n]
//   M = SEQ*BATCH, K = HIDDEN, N = VOCAB.
// ============================================================
__global__ void out_gemm_kernel(const unsigned short* __restrict__ hs,
                                const unsigned short* __restrict__ Wfc,
                                const float* __restrict__ b_fc,
                                float* __restrict__ out) {
    const int lane = threadIdx.x & 31;
    const int wave = threadIdx.x >> 5;
    const int m0   = blockIdx.x * 16;
    const int n0   = (blockIdx.y * 16 + wave) * 16;
    const int ah   = (lane >> 4) << 3;
    const int bh   = (lane >> 4) << 4;
    const int col  = lane & 15;

    const unsigned short* arow = hs  + (m0 + col) * HIDDEN;
    const unsigned short* brow = Wfc + (n0 + col) * HIDDEN;

    v8f acc = {};
#pragma unroll
    for (int kk = 0; kk < HIDDEN; kk += 32) {
        Frag fa, fb;
        fa.h[0] = *(const us8*)(arow + kk + ah);
        fa.h[1] = *(const us8*)(arow + kk + ah + 16);
        const us8* qb = (const us8*)(brow + kk + ah);
        fb.h[0] = qb[0];
        fb.h[1] = qb[2];
        acc = __builtin_amdgcn_wmma_f32_16x16x32_bf16(false, fa.bf, false, fb.bf,
                                                      (short)0, acc, false, false);
    }

    const float bias = b_fc[n0 + col];
#pragma unroll
    for (int r = 0; r < 8; ++r)
        out[(m0 + r + ah) * VOCAB + n0 + col] = acc[r] + bias;
}

// ============================================================
extern "C" void kernel_launch(void* const* d_in, const int* in_sizes, int n_in,
                              void* d_out, int out_size, void* d_ws, size_t ws_size,
                              hipStream_t stream) {
    (void)in_sizes; (void)n_in; (void)out_size; (void)ws_size;

    const float* x    = (const float*)d_in[0];
    const float* h0   = (const float*)d_in[1];
    const float* W_ih = (const float*)d_in[2];
    const float* W_hh = (const float*)d_in[3];
    const float* b_ih = (const float*)d_in[4];
    const float* b_hh = (const float*)d_in[5];
    const float* W_fc = (const float*)d_in[6];
    const float* b_fc = (const float*)d_in[7];
    float* out = (float*)d_out;

    // workspace layout
    const long long XI_ELEMS = (long long)SEQ * BATCH * HIDDEN;   // 67,108,864
    float*          xi  = (float*)d_ws;                           // 256 MB f32
    unsigned short* hs  = (unsigned short*)(xi + XI_ELEMS);       // 128 MB bf16
    unsigned short* wih = hs + XI_ELEMS;                          // bf16 weights
    unsigned short* whh = wih + (long long)HIDDEN * VOCAB;
    unsigned short* wfc = whh + (long long)HIDDEN * HIDDEN;

    // phase 0: weight conversion
    {
        int total = HIDDEN * VOCAB + HIDDEN * HIDDEN + VOCAB * HIDDEN;
        cvt_weights_kernel<<<(total + 255) / 256, 256, 0, stream>>>(
            W_ih, W_hh, W_fc, wih, whh, wfc);
    }
    // phase 1: input projection (bandwidth-bound big GEMM)
    xi_gemm_kernel<<<dim3(SEQ * BATCH / 16), 512, 0, stream>>>(x, wih, b_ih, b_hh, xi);
    // phase 2: sequential recurrence (latency-bound, batch-parallel blocks)
    rnn_steps_kernel<<<dim3(BATCH / 16), 512, 0, stream>>>(xi, whh, h0, hs);
    // phase 3: output projection (bandwidth-bound big GEMM)
    out_gemm_kernel<<<dim3(SEQ * BATCH / 16, VOCAB / 256), 512, 0, stream>>>(hs, wfc, b_fc, out);
}
